// DnnDependencyParser_15247133901515
// MI455X (gfx1250) — compile-verified, matrix operand order checked by reading, not run on hardware
//
#include <hip/hip_runtime.h>
#include <hip/hip_bf16.h>

typedef __attribute__((ext_vector_type(16))) _Float16 v16h;
typedef __attribute__((ext_vector_type(8)))  float    v8f;

#define T_SEQ 1024
#define HID   512
#define GATES 2048   // 4*HID
#define DMODEL 1024  // 2*HID
#define K0PAD 128    // 125 padded
#define MLPPAD 64    // 50 padded

// ---------------------------------------------------------------------------
// Embedding gather + concat -> f16 [T][128] (cols 0..99 word, 100..124 pos, rest 0)
// ---------------------------------------------------------------------------
__global__ __launch_bounds__(128) void embed_kernel(
    const int* __restrict__ word_idx, const int* __restrict__ pos_idx,
    const float* __restrict__ Ew, const float* __restrict__ Ep,
    _Float16* __restrict__ xf16)
{
    int t = blockIdx.x;
    int k = threadIdx.x;
    float v = 0.0f;
    if (k < 100)       v = Ew[(size_t)word_idx[t] * 100 + k];
    else if (k < 125)  v = Ep[(size_t)pos_idx[t] * 25 + (k - 100)];
    xf16[(size_t)t * K0PAD + k] = (_Float16)v;
}

// ---------------------------------------------------------------------------
// Generic f32 -> f16 convert with zero padding: dst[Nd][Kd] from src[Ns][Ks]
// ---------------------------------------------------------------------------
__global__ __launch_bounds__(256) void cvt_pad_f16(
    const float* __restrict__ src, _Float16* __restrict__ dst,
    int Nd, int Kd, int Ns, int Ks)
{
    int i = blockIdx.x * 256 + threadIdx.x;
    if (i >= Nd * Kd) return;
    int n = i / Kd, k = i - n * Kd;
    float v = (n < Ns && k < Ks) ? src[(size_t)n * Ks + k] : 0.0f;
    dst[i] = (_Float16)v;
}

// ---------------------------------------------------------------------------
// Transpose Whh [2048][512] f32 -> WhhT [512][2048] f16 (recurrence streams
// rows of WhhT coalesced each step)
// ---------------------------------------------------------------------------
__global__ __launch_bounds__(256) void transpose_whh_f16(
    const float* __restrict__ src, _Float16* __restrict__ dst)
{
    int i = blockIdx.x * 256 + threadIdx.x;   // over 2048*512
    if (i >= GATES * HID) return;
    int n = i / HID, k = i - n * HID;         // src[n][k]
    dst[(size_t)k * GATES + n] = (_Float16)src[i];
}

// ---------------------------------------------------------------------------
// WMMA GEMM: C[M][N] = A[M][K](f16) * W[N][K](f16)^T + bias
// Block = 256 threads (8 waves). Block tile: 128 M-rows x 64 N-cols.
//   - W slab (64 N-rows x 32 K = 4KB) staged into LDS by async copy,
//     double-buffered, shared by all 8 waves  -> 8x less W traffic from L2.
//   - Each wave owns one 16-row M-tile, loads one A fragment per K-step,
//     and issues 4 wmma against 4 B fragments read from LDS.
// Requires M % 128 == 0, N % 64 == 0, K % 32 == 0.
// ---------------------------------------------------------------------------
__global__ __launch_bounds__(256) void wmma_gemm_nt(
    const _Float16* __restrict__ A, const _Float16* __restrict__ W,
    const float* __restrict__ bias, int nbias,
    float* __restrict__ C, int M, int N, int K)
{
    __shared__ _Float16 wtile[2][64 * 32];    // two 4KB W slabs

    const int tid  = threadIdx.x;
    const int wave = tid >> 5;
    const int lane = tid & 31;
    const int half = lane >> 4;               // K sub-chunk carried by this lane
    const int lr   = lane & 15;

    const int nBlocks = N >> 6;
    const int mb = (blockIdx.x / nBlocks) * 128;
    const int nb = (blockIdx.x % nBlocks) * 64;

    // async staging assignment: thread copies one 16B chunk of the W slab:
    // W row (nb + tid/4), K chunk (tid%4)*8 halves
    const _Float16* wsrc = W + (size_t)(nb + (tid >> 2)) * K + (tid & 3) * 8;
    const unsigned  ldsByte = (unsigned)(tid * 16);
    const unsigned  ldsBase0 = (unsigned)(uintptr_t)(&wtile[0][0]);
    const unsigned  ldsBase1 = (unsigned)(uintptr_t)(&wtile[1][0]);

    const _Float16* arow = A + (size_t)(mb + wave * 16 + lr) * K + half * 8;

    v8f acc[4] = {};

    // prologue: stage kt = 0 into buffer 0
    asm volatile("global_load_async_to_lds_b128 %0, %1, off"
                 :: "v"(ldsBase0 + ldsByte), "v"(wsrc) : "memory");

    for (int kt = 0; kt < K; kt += 32) {
        const int cur = (kt >> 5) & 1;
        if (kt + 32 < K) {
            const unsigned dst = (cur ? ldsBase0 : ldsBase1) + ldsByte;
            asm volatile("global_load_async_to_lds_b128 %0, %1, off"
                         :: "v"(dst), "v"(wsrc + kt + 32) : "memory");
            // FIFO per-thread completion: <=1 outstanding means slab kt is done
            asm volatile("s_wait_asynccnt 0x1" ::: "memory");
        } else {
            asm volatile("s_wait_asynccnt 0x0" ::: "memory");
        }
        __syncthreads();                       // slab kt visible to all waves

        v16h a;
        ((uint4*)&a)[0] = *(const uint4*)(arow + kt);        // K = half*8 + [0,8)
        ((uint4*)&a)[1] = *(const uint4*)(arow + kt + 16);   // K = half*8+16 + [0,8)
        __builtin_prefetch(arow + kt + 128);                 // global_prefetch_b8

        const _Float16* wl = &wtile[cur][0];
#pragma unroll
        for (int j = 0; j < 4; ++j) {
            const _Float16* brow = wl + (j * 16 + lr) * 32 + half * 8;
            v16h b;
            ((uint4*)&b)[0] = *(const uint4*)(brow);         // ds_load_b128
            ((uint4*)&b)[1] = *(const uint4*)(brow + 16);
            acc[j] = __builtin_amdgcn_wmma_f32_16x16x32_f16(
                         false, a, false, b, (short)0, acc[j], false, false);
        }
        __syncthreads();                       // all reads done before overwrite
    }

    const int mrow = mb + wave * 16;
#pragma unroll
    for (int j = 0; j < 4; ++j) {
        const int col = nb + j * 16 + lr;
        const float bv = (bias != nullptr && col < nbias) ? bias[col] : 0.0f;
#pragma unroll
        for (int r = 0; r < 8; ++r) {          // C/D layout: VGPR r -> M=r(+8)
            C[(size_t)(mrow + r + half * 8) * N + col] = acc[j][r] + bv;
        }
    }
}

// ---------------------------------------------------------------------------
// One bidirectional LSTM layer. grid.x = 2 (0=fwd, 1=bwd), 1024 threads.
// pre: [T][2048] precomputed x@Wih^T + b.  whhT: [512][2048] f16.
// Writes h (f16) into out16[t][colOff + j].
// ---------------------------------------------------------------------------
__global__ __launch_bounds__(1024) void lstm_layer(
    const float* __restrict__ pre_f, const float* __restrict__ pre_b,
    const _Float16* __restrict__ whhT_f, const _Float16* __restrict__ whhT_b,
    _Float16* __restrict__ out16, int T)
{
    __shared__ float h[HID];
    __shared__ float c[HID];
    __shared__ float z[GATES];

    const int tid = threadIdx.x;
    const bool rev = (blockIdx.x == 1);
    const float*    pre = rev ? pre_b : pre_f;
    const _Float16* wT  = rev ? whhT_b : whhT_f;
    const int colOff    = rev ? HID : 0;

    if (tid < HID) { h[tid] = 0.0f; c[tid] = 0.0f; }
    __syncthreads();

    for (int step = 0; step < T; ++step) {
        const int t = rev ? (T - 1 - step) : step;
        const float* prow = pre + (size_t)t * GATES;
        float a0 = prow[tid];
        float a1 = prow[tid + 1024];
#pragma unroll 4
        for (int k = 0; k < HID; ++k) {
            const float hk = h[k];
            const _Float16* wr = wT + (size_t)k * GATES;
            a0 = fmaf((float)wr[tid],        hk, a0);
            a1 = fmaf((float)wr[tid + 1024], hk, a1);
        }
        z[tid] = a0;
        z[tid + 1024] = a1;
        __syncthreads();
        if (tid < HID) {
            const float zi = z[tid];
            const float zf = z[HID + tid];
            const float zg = z[2 * HID + tid];
            const float zo = z[3 * HID + tid];
            const float ig = 1.0f / (1.0f + __expf(-zi));
            const float fg = 1.0f / (1.0f + __expf(-zf));
            const float gg = tanhf(zg);
            const float og = 1.0f / (1.0f + __expf(-zo));
            const float cn = fg * c[tid] + ig * gg;
            const float hn = og * tanhf(cn);
            c[tid] = cn;
            h[tid] = hn;
            out16[(size_t)t * DMODEL + colOff + tid] = (_Float16)hn;
        }
        __syncthreads();
    }
}

// ---------------------------------------------------------------------------
// scores[h][m] = w2 . relu(heads[h] + mods[m]) + b2 ; diagonal = -inf
// ---------------------------------------------------------------------------
__global__ __launch_bounds__(256) void scores_kernel(
    const float* __restrict__ heads, const float* __restrict__ mods,
    const float* __restrict__ w2, const float* __restrict__ b2,
    float* __restrict__ out, int T)
{
    __shared__ float hrow[50];
    __shared__ float w2s[50];
    const int hIdx = blockIdx.x;
    const int m = blockIdx.y * 256 + threadIdx.x;
    if (threadIdx.x < 50) {
        hrow[threadIdx.x] = heads[(size_t)hIdx * MLPPAD + threadIdx.x];
        w2s[threadIdx.x]  = w2[threadIdx.x];
    }
    __syncthreads();
    float s = b2[0];
    const float* mrow = mods + (size_t)m * MLPPAD;
#pragma unroll 5
    for (int k = 0; k < 50; ++k) {
        const float v = hrow[k] + mrow[k];
        s = fmaf(w2s[k], fmaxf(v, 0.0f), s);
    }
    if (m == hIdx) s = -__builtin_inff();
    out[(size_t)hIdx * T + m] = s;
}

// ---------------------------------------------------------------------------
extern "C" void kernel_launch(void* const* d_in, const int* in_sizes, int n_in,
                              void* d_out, int out_size, void* d_ws, size_t ws_size,
                              hipStream_t stream) {
    (void)in_sizes; (void)n_in; (void)out_size; (void)ws_size;
    const int*   word_idx = (const int*)d_in[0];
    const int*   pos_idx  = (const int*)d_in[1];
    const float* Ew    = (const float*)d_in[2];
    const float* Ep    = (const float*)d_in[3];
    const float* Wih0f = (const float*)d_in[4];
    const float* Whh0f = (const float*)d_in[5];
    const float* b0f   = (const float*)d_in[6];
    const float* Wih0b = (const float*)d_in[7];
    const float* Whh0b = (const float*)d_in[8];
    const float* b0b   = (const float*)d_in[9];
    const float* Wih1f = (const float*)d_in[10];
    const float* Whh1f = (const float*)d_in[11];
    const float* b1f   = (const float*)d_in[12];
    const float* Wih1b = (const float*)d_in[13];
    const float* Whh1b = (const float*)d_in[14];
    const float* b1b   = (const float*)d_in[15];
    const float* Wh    = (const float*)d_in[16];
    const float* bh    = (const float*)d_in[17];
    const float* Wm    = (const float*)d_in[18];
    const float* w2    = (const float*)d_in[19];
    const float* b2    = (const float*)d_in[20];
    float* out = (float*)d_out;

    char* ws = (char*)d_ws;
    size_t off = 0;
    auto alloc = [&](size_t bytes) -> char* {
        char* p = ws + off;
        off += (bytes + 255) & ~(size_t)255;
        return p;
    };
    _Float16* xf16   = (_Float16*)alloc((size_t)T_SEQ * K0PAD * 2);
    _Float16* w0f16f = (_Float16*)alloc((size_t)GATES * K0PAD * 2);
    _Float16* w0f16b = (_Float16*)alloc((size_t)GATES * K0PAD * 2);
    _Float16* w1f16f = (_Float16*)alloc((size_t)GATES * DMODEL * 2);
    _Float16* w1f16b = (_Float16*)alloc((size_t)GATES * DMODEL * 2);
    _Float16* wh16   = (_Float16*)alloc((size_t)MLPPAD * DMODEL * 2);
    _Float16* wm16   = (_Float16*)alloc((size_t)MLPPAD * DMODEL * 2);
    _Float16* whhT0f = (_Float16*)alloc((size_t)HID * GATES * 2);
    _Float16* whhT0b = (_Float16*)alloc((size_t)HID * GATES * 2);
    _Float16* whhT1f = (_Float16*)alloc((size_t)HID * GATES * 2);
    _Float16* whhT1b = (_Float16*)alloc((size_t)HID * GATES * 2);
    float*    preA   = (float*)alloc((size_t)T_SEQ * GATES * 4);   // reused layer0/1
    float*    preB   = (float*)alloc((size_t)T_SEQ * GATES * 4);
    _Float16* h0f16  = (_Float16*)alloc((size_t)T_SEQ * DMODEL * 2);
    _Float16* out16  = (_Float16*)alloc((size_t)T_SEQ * DMODEL * 2);
    float*    headsB = (float*)alloc((size_t)T_SEQ * MLPPAD * 4);
    float*    modsB  = (float*)alloc((size_t)T_SEQ * MLPPAD * 4);

    // 1) embeddings -> f16
    embed_kernel<<<T_SEQ, 128, 0, stream>>>(word_idx, pos_idx, Ew, Ep, xf16);

    // 2) weight precision conversion / padding
    cvt_pad_f16<<<(GATES * K0PAD + 255) / 256, 256, 0, stream>>>(Wih0f, w0f16f, GATES, K0PAD, GATES, 125);
    cvt_pad_f16<<<(GATES * K0PAD + 255) / 256, 256, 0, stream>>>(Wih0b, w0f16b, GATES, K0PAD, GATES, 125);
    cvt_pad_f16<<<(GATES * DMODEL + 255) / 256, 256, 0, stream>>>(Wih1f, w1f16f, GATES, DMODEL, GATES, DMODEL);
    cvt_pad_f16<<<(GATES * DMODEL + 255) / 256, 256, 0, stream>>>(Wih1b, w1f16b, GATES, DMODEL, GATES, DMODEL);
    cvt_pad_f16<<<(MLPPAD * DMODEL + 255) / 256, 256, 0, stream>>>(Wh, wh16, MLPPAD, DMODEL, 50, DMODEL);
    cvt_pad_f16<<<(MLPPAD * DMODEL + 255) / 256, 256, 0, stream>>>(Wm, wm16, MLPPAD, DMODEL, 50, DMODEL);
    transpose_whh_f16<<<(GATES * HID + 255) / 256, 256, 0, stream>>>(Whh0f, whhT0f);
    transpose_whh_f16<<<(GATES * HID + 255) / 256, 256, 0, stream>>>(Whh0b, whhT0b);
    transpose_whh_f16<<<(GATES * HID + 255) / 256, 256, 0, stream>>>(Whh1f, whhT1f);
    transpose_whh_f16<<<(GATES * HID + 255) / 256, 256, 0, stream>>>(Whh1b, whhT1b);

    // 3) layer-0 input projections (WMMA): [1024,128] x [2048,128]^T
    const int blkPre = (T_SEQ / 128) * (GATES / 64);   // 8 * 32 = 256 blocks
    wmma_gemm_nt<<<blkPre, 256, 0, stream>>>(xf16, w0f16f, b0f, GATES, preA, T_SEQ, GATES, K0PAD);
    wmma_gemm_nt<<<blkPre, 256, 0, stream>>>(xf16, w0f16b, b0b, GATES, preB, T_SEQ, GATES, K0PAD);

    // 4) layer-0 recurrence (fwd + bwd concurrently)
    lstm_layer<<<2, 1024, 0, stream>>>(preA, preB, whhT0f, whhT0b, h0f16, T_SEQ);

    // 5) layer-1 input projections (WMMA): [1024,1024] x [2048,1024]^T
    wmma_gemm_nt<<<blkPre, 256, 0, stream>>>(h0f16, w1f16f, b1f, GATES, preA, T_SEQ, GATES, DMODEL);
    wmma_gemm_nt<<<blkPre, 256, 0, stream>>>(h0f16, w1f16b, b1b, GATES, preB, T_SEQ, GATES, DMODEL);

    // 6) layer-1 recurrence
    lstm_layer<<<2, 1024, 0, stream>>>(preA, preB, whhT1f, whhT1b, out16, T_SEQ);

    // 7) heads / mods projections (WMMA, N padded 50->64)
    const int blkHM = (T_SEQ / 128) * (MLPPAD / 64);   // 8 blocks
    wmma_gemm_nt<<<blkHM, 256, 0, stream>>>(out16, wh16, bh, 50, headsB, T_SEQ, MLPPAD, DMODEL);
    wmma_gemm_nt<<<blkHM, 256, 0, stream>>>(out16, wm16, nullptr, 0, modsB, T_SEQ, MLPPAD, DMODEL);

    // 8) pairwise biaffine scores
    scores_kernel<<<dim3(T_SEQ, T_SEQ / 256), 256, 0, stream>>>(headsB, modsB, w2, b2, out, T_SEQ);
}
